// WanCrossAttnCachingLinear_46729244180735
// MI455X (gfx1250) — compile-verified, matrix operand order checked by reading, not run on hardware
//
#include <hip/hip_runtime.h>

typedef __attribute__((ext_vector_type(16))) __bf16 v16bf;
typedef __attribute__((ext_vector_type(8)))  float  v8f;
typedef unsigned int u32x4 __attribute__((ext_vector_type(4)));
typedef int          i32x8 __attribute__((ext_vector_type(8)));
typedef int          i32x4 __attribute__((ext_vector_type(4)));

#define M_TOTAL 1024
#define K_TOTAL 5120
#define N_TOTAL 5120
#define HEAD    128
#define MT      64
#define KT      32

// Issue one 2D TDM tile load: tile_rows x KT f32 elements, row stride = stride_elems.
// D# layout per CDNA5 ISA 8.3/8.4: group0 = {flags, lds_addr, global_addr, type},
// group1 = {data_size, tensor dims, tile dims, strides}; groups 2/3 unused (2D).
__device__ __forceinline__ void tdm_load_2d(unsigned lds_addr, const float* gptr,
                                            unsigned tile_rows, unsigned stride_elems)
{
    unsigned long long ga = (unsigned long long)(uintptr_t)gptr;
    u32x4 g0;
    g0[0] = 1u;                                      // count=1 (valid user descriptor)
    g0[1] = lds_addr;                                // LDS byte address
    g0[2] = (unsigned)(ga & 0xffffffffu);            // global_addr[31:0]
    g0[3] = (unsigned)((ga >> 32) & 0x01ffffffu)     // global_addr[56:32]
          | 0x80000000u;                             // type=2 ("image") in bits 127:126
    i32x8 g1;
    g1[0] = (int)(2u << 16);                         // data_size = 2 -> 4 bytes
    g1[1] = (int)((stride_elems & 0xffffu) << 16);   // tensor_dim0[15:0]
    g1[2] = (int)((stride_elems >> 16)               // tensor_dim0[31:16]
          | (tile_rows << 16));                      // tensor_dim1[15:0]
    g1[3] = (int)(KT << 16);                         // tensor_dim1 hi=0 | tile_dim0=KT
    g1[4] = (int)tile_rows;                          // tile_dim1 | tile_dim2=0
    g1[5] = (int)stride_elems;                       // tensor_dim0_stride[31:0]
    g1[6] = 0;                                       // stride hi | tensor_dim1_stride lo
    g1[7] = 0;
    i32x4 z4 = {0, 0, 0, 0};
    i32x8 z8 = {0, 0, 0, 0, 0, 0, 0, 0};
    // amdgpu-toolchain (clang-23) 6-arg form: (g0, g1, g2, g3, g4, cpol)
    __builtin_amdgcn_tensor_load_to_lds(g0, g1, z4, z4, z8, 0);
}

// Build a 16-element bf16 WMMA fragment from an f32 LDS row (32 elements):
// per-lane K = {kh..kh+7, 16+kh..16+kh+7} -> two contiguous 32B runs
// -> 4x ds_load_b128 + packed v_cvt_pk_bf16_f32.
__device__ __forceinline__ v16bf frag_from_f32row(const float* row, int kh)
{
    const float4* rp = reinterpret_cast<const float4*>(row);
    const int q = kh >> 2;                // 0 or 2
    const float4 c0 = rp[q];              // K kh..kh+3
    const float4 c1 = rp[q + 1];          // K kh+4..kh+7
    const float4 c2 = rp[q + 4];          // K 16+kh..16+kh+3
    const float4 c3 = rp[q + 5];          // K 16+kh+4..16+kh+7
    v16bf f;
    f[0]  = (__bf16)c0.x; f[1]  = (__bf16)c0.y; f[2]  = (__bf16)c0.z; f[3]  = (__bf16)c0.w;
    f[4]  = (__bf16)c1.x; f[5]  = (__bf16)c1.y; f[6]  = (__bf16)c1.z; f[7]  = (__bf16)c1.w;
    f[8]  = (__bf16)c2.x; f[9]  = (__bf16)c2.y; f[10] = (__bf16)c2.z; f[11] = (__bf16)c2.w;
    f[12] = (__bf16)c3.x; f[13] = (__bf16)c3.y; f[14] = (__bf16)c3.z; f[15] = (__bf16)c3.w;
    return f;
}

__launch_bounds__(256, 1)
__global__ void gemm_hadamard_quant(const float* __restrict__ x,
                                    const float* __restrict__ W,
                                    const float* __restrict__ bias,
                                    float* __restrict__ out)
{
    // Double-buffered f32 TDM stages; epilogue tile aliases them (dead after K-loop).
    __shared__ union {
        struct { float A[2][MT][KT]; float B[2][HEAD][KT]; } st;  // 48 KB
        float S[MT][HEAD];                                        // 32 KB
    } u;
    __shared__ int sPerm[HEAD];

    const int tid  = threadIdx.x;
    const int lane = tid & 31;
    const int wv   = tid >> 5;            // wave 0..7
    const int wm   = wv & 3;              // M sub-tile (rows wm*16..)
    const int wn   = wv >> 2;             // N half (cols wn*64..)
    const int m0   = blockIdx.x * MT;
    const int head = blockIdx.y;
    const int e0   = head * HEAD;

    // ---- Mobius permutation (mu==1 indices, then mu==0, then mu==-1) ----
    if (tid == 0) {
        int  mu[HEAD + 1];
        bool comp[HEAD + 1];
        for (int i = 0; i <= HEAD; ++i) { mu[i] = 1; comp[i] = false; }
        for (int p = 2; p <= HEAD; ++p) {
            if (!comp[p]) {
                for (int q = 2 * p; q <= HEAD; q += p) comp[q] = true;
                for (int q = p;     q <= HEAD; q += p) mu[q] = -mu[q];
                int sq = p * p;
                if (sq <= HEAD)
                    for (int q = sq; q <= HEAD; q += sq) mu[q] = 0;
            }
        }
        int idx = 0;
        for (int v = 1; v >= -1; --v)
            for (int n = 1; n <= HEAD; ++n)
                if (mu[n] == v) sPerm[idx++] = n - 1;
    }

    v8f acc[4] = {};

    // =================== GEMM: C = x @ W^T (TDM-staged, bf16 WMMA, f32 acc) =========
    if (wv == 0) {   // prologue: fill buffer 0
        tdm_load_2d((unsigned)(uintptr_t)&u.st.A[0][0][0],
                    x + (size_t)m0 * K_TOTAL, MT, K_TOTAL);
        tdm_load_2d((unsigned)(uintptr_t)&u.st.B[0][0][0],
                    W + (size_t)e0 * K_TOTAL, HEAD, K_TOTAL);
    }

    for (int kk = 0, step = 0; kk < K_TOTAL; kk += KT, ++step) {
        const int buf = step & 1;
        if (wv == 0)
            __builtin_amdgcn_s_wait_tensorcnt(0);   // current buffer landed in LDS
        __syncthreads();
        if (wv == 0 && (kk + KT) < K_TOTAL) {        // prefetch next K tile via TDM
            tdm_load_2d((unsigned)(uintptr_t)&u.st.A[buf ^ 1][0][0],
                        x + (size_t)m0 * K_TOTAL + kk + KT, MT, K_TOTAL);
            tdm_load_2d((unsigned)(uintptr_t)&u.st.B[buf ^ 1][0][0],
                        W + (size_t)e0 * K_TOTAL + kk + KT, HEAD, K_TOTAL);
        }

        // A fragment: lanes 0-15 -> M=lane, K 0..7/16..23 ; lanes 16-31 -> K 8..15/24..31
        const int arow = wm * 16 + (lane & 15);
        const int kh   = (lane >> 4) * 8;
        const v16bf a = frag_from_f32row(&u.st.A[buf][arow][0], kh);
        #pragma unroll
        for (int t = 0; t < 4; ++t) {
            const int bcol = (wn * 4 + t) * 16 + (lane & 15);
            const v16bf bm = frag_from_f32row(&u.st.B[buf][bcol][0], kh);
            acc[t] = __builtin_amdgcn_wmma_f32_16x16x32_bf16(
                false, a, false, bm, (short)0, acc[t], false, false);
        }
        __syncthreads();
    }

    // =================== write accumulators (+bias) into S ===================
    {
        const int nrel  = lane & 15;
        const int mbase = (lane >> 4) * 8;
        #pragma unroll
        for (int t = 0; t < 4; ++t) {
            const int col = (wn * 4 + t) * 16 + nrel;
            const float bv = bias[e0 + col];
            #pragma unroll
            for (int j = 0; j < 8; ++j)
                u.S[wm * 16 + mbase + j][col] = acc[t][j] + bv;
        }
    }
    __syncthreads();

    // =================== FWHT #1 (z = y_row @ H, Sylvester order) ===================
    for (int stage = 0; stage < 7; ++stage) {
        const int half = 1 << stage;
        #pragma unroll
        for (int it = 0; it < (MT * 64) / 256; ++it) {
            const int bf  = tid + it * 256;
            const int row = bf >> 6;
            const int j   = bf & 63;
            const int lo  = ((j >> stage) << (stage + 1)) | (j & (half - 1));
            const int hi  = lo + half;
            const float p = u.S[row][lo];
            const float q = u.S[row][hi];
            u.S[row][lo] = p + q;
            u.S[row][hi] = p - q;
        }
        __syncthreads();
    }

    // =========== banded quant-dequant (bands of 32, qmax 15/7/7/3) ===========
    {
        const float qmaxs[4] = {15.f, 7.f, 7.f, 3.f};
        #pragma unroll
        for (int r8 = 0; r8 < 8; ++r8) {
            const int row = wv * 8 + r8;
            float vals[4];
            int   eidx[4];
            #pragma unroll
            for (int t = 0; t < 4; ++t) {
                eidx[t] = sPerm[t * 32 + lane];   // band t, element = lane
                vals[t] = u.S[row][eidx[t]];
            }
            #pragma unroll
            for (int t = 0; t < 4; ++t) {
                float am = fabsf(vals[t]);
                #pragma unroll
                for (int off = 16; off >= 1; off >>= 1)
                    am = fmaxf(am, __shfl_xor(am, off, 32));
                const float sc = (am > 0.f) ? (am / qmaxs[t]) : 1.f;
                float q = rintf(vals[t] / sc);
                q = fminf(fmaxf(q, -qmaxs[t]), qmaxs[t]);
                u.S[row][eidx[t]] = q * sc;       // perm then inv-perm round-trips
            }
        }
    }
    __syncthreads();

    // =================== FWHT #2 ===================
    for (int stage = 0; stage < 7; ++stage) {
        const int half = 1 << stage;
        #pragma unroll
        for (int it = 0; it < (MT * 64) / 256; ++it) {
            const int bf  = tid + it * 256;
            const int row = bf >> 6;
            const int j   = bf & 63;
            const int lo  = ((j >> stage) << (stage + 1)) | (j & (half - 1));
            const int hi  = lo + half;
            const float p = u.S[row][lo];
            const float q = u.S[row][hi];
            u.S[row][lo] = p + q;
            u.S[row][hi] = p - q;
        }
        __syncthreads();
    }

    // =================== scale 1/128, NaN scrub, clip, store ===================
    #pragma unroll
    for (int it = 0; it < (MT * HEAD) / 256; ++it) {
        const int idx = tid + it * 256;
        const int row = idx >> 7;
        const int e   = idx & 127;
        float v = u.S[row][e] * (1.0f / 128.0f);
        v = (v == v) ? v : 0.0f;
        v = fminf(fmaxf(v, -65504.f), 65504.f);
        out[(size_t)(m0 + row) * N_TOTAL + e0 + e] = v;
    }
}

extern "C" void kernel_launch(void* const* d_in, const int* in_sizes, int n_in,
                              void* d_out, int out_size, void* d_ws, size_t ws_size,
                              hipStream_t stream) {
    (void)in_sizes; (void)n_in; (void)out_size; (void)d_ws; (void)ws_size;
    const float* x = (const float*)d_in[0];
    const float* W = (const float*)d_in[1];
    const float* b = (const float*)d_in[2];
    float* out = (float*)d_out;
    dim3 grid(M_TOTAL / MT, N_TOTAL / HEAD);
    gemm_hadamard_quant<<<grid, 256, 0, stream>>>(x, W, b, out);
}
